// LlamaAttention_55027120996966
// MI455X (gfx1250) — matrix operand order repaired and checked by reading, not compiled, not run on hardware
//
#include <hip/hip_runtime.h>
#include <hip/hip_bf16.h>

// ---------------------------------------------------------------------------
// LlamaAttention (B=1, S=2048, HID=2048, NH=32, NKV=8, HD=64, WINDOW=1024)
// All GEMM-like stages via v_wmma_f32_16x16x32_bf16 (fp32 accumulate).
// ---------------------------------------------------------------------------

typedef __bf16 v16bf __attribute__((ext_vector_type(16)));
typedef __bf16 v8bf  __attribute__((ext_vector_type(8)));
typedef float  v8f   __attribute__((ext_vector_type(8)));

#define S_LEN    2048
#define HID_DIM  2048
#define N_HEADS  32
#define N_KV     8
#define HEAD_DIM 64
#define KV_DIM   (N_KV * HEAD_DIM)   // 512
#define WIN      1024
#define LOG2_ROPE_BASE 18.931568569324174f   // log2(500000)

static __device__ __forceinline__ unsigned short f2bf(float f) {
  unsigned u = __builtin_bit_cast(unsigned, f);
  u += 0x7FFFu + ((u >> 16) & 1u);     // round-to-nearest-even
  return (unsigned short)(u >> 16);
}

static __device__ __forceinline__ v8f wmma_bf16(v16bf a, v16bf b, v8f c) {
  return __builtin_amdgcn_wmma_f32_16x16x32_bf16(false, a, false, b,
                                                 (short)0, c, false, false);
}

union V16U { v16bf v; v8bf h[2]; };

static __device__ __forceinline__ v16bf ld16(const unsigned short* p0,
                                             const unsigned short* p1) {
  V16U u;
  u.h[0] = *(const v8bf*)p0;
  u.h[1] = *(const v8bf*)p1;
  return u.v;
}

static __device__ __forceinline__ float rmax16(float v) {
  v = fmaxf(v, __shfl_xor(v, 1, 32));
  v = fmaxf(v, __shfl_xor(v, 2, 32));
  v = fmaxf(v, __shfl_xor(v, 4, 32));
  v = fmaxf(v, __shfl_xor(v, 8, 32));
  return v;
}
static __device__ __forceinline__ float rsum16(float v) {
  v += __shfl_xor(v, 1, 32);
  v += __shfl_xor(v, 2, 32);
  v += __shfl_xor(v, 4, 32);
  v += __shfl_xor(v, 8, 32);
  return v;
}

// ------------------------------ conversions --------------------------------

__global__ __launch_bounds__(256) void k_f32_to_bf16(const float* __restrict__ in,
                                                     unsigned short* __restrict__ out,
                                                     int n) {
  int i = blockIdx.x * 256 + threadIdx.x;
  if (i < n) out[i] = f2bf(in[i]);
}

// W: [K][N] fp32  ->  Wt: [N][K] bf16
__global__ __launch_bounds__(256) void k_transpose_bf16(const float* __restrict__ W,
                                                        unsigned short* __restrict__ Wt,
                                                        int K, int N) {
  int i = blockIdx.x * 256 + threadIdx.x;
  if (i >= N * K) return;
  int n = i / K, k = i - n * K;
  Wt[(size_t)n * K + k] = f2bf(W[(size_t)k * N + n]);
}

// ------------------------------ WMMA GEMM ----------------------------------
// C[M][N] = A[M][K](bf16) @ Bt[N][K](bf16)^T, wave tile 16x64, block 128x64.
// MODE 0: fp32 row-major out.  MODE 1: fused RoPE -> bf16 row-major (N tile
// == one 64-wide head).  MODE 2: bf16 out transposed per-kv-head [kv][hd][M].

template <int MODE>
__global__ __launch_bounds__(256) void k_gemm(const unsigned short* __restrict__ A,
                                              const unsigned short* __restrict__ Bt,
                                              float* __restrict__ outf,
                                              unsigned short* __restrict__ outb,
                                              int M, int N, int K) {
  const int wave = threadIdx.x >> 5;
  const int lane = threadIdx.x & 31;
  const int ln   = lane & 15;
  const int hh   = lane >> 4;
  const int m0   = blockIdx.x * 128 + wave * 16;
  const int n0   = blockIdx.y * 64;

  const v8f z8 = {0.f, 0.f, 0.f, 0.f, 0.f, 0.f, 0.f, 0.f};
  v8f acc[4] = {z8, z8, z8, z8};

  const unsigned short* arow = A + (size_t)(m0 + ln) * K;

  for (int k0 = 0; k0 < K; k0 += 32) {
    const int ab = k0 + (hh ? 16 : 0);
    v16bf af = ld16(arow + ab, arow + ab + 8);           // A 16x32 fragment
    __builtin_prefetch(arow + k0 + 512, 0, 0);
#pragma unroll
    for (int f = 0; f < 4; ++f) {
      const unsigned short* brow =
          Bt + (size_t)(n0 + f * 16 + ln) * K + k0 + hh * 16;
      v16bf bfr = ld16(brow, brow + 8);                  // B 32x16 fragment
      acc[f] = wmma_bf16(af, bfr, acc[f]);
    }
  }

#pragma unroll
  for (int f = 0; f < 4; ++f) {
#pragma unroll
    for (int i = 0; i < 8; ++i) {
      const int m = m0 + i + 8 * hh;     // C layout: lanes<16 -> M=i, else M=i+8
      const int n = n0 + f * 16 + ln;
      const float val = acc[f][i];
      if (MODE == 0) {
        outf[(size_t)m * N + n] = val;
      } else if (MODE == 1) {
        // RoPE: pair (hd, hd^32) lives in fragment f^2, same lane & i.
        const float xp = acc[f ^ 2][i];
        const int hd = f * 16 + ln;      // 0..63 within this head
        const int h2 = hd & 31;
        const float ang =
            (float)m * exp2f(-(float)h2 * (LOG2_ROPE_BASE / 32.f));
        float sn, cs;
        __sincosf(ang, &sn, &cs);
        const float rot = (hd < 32) ? -xp : xp;
        outb[(size_t)m * N + n] = f2bf(val * cs + rot * sn);
      } else {  // MODE 2: V transposed [kv][hd][M]
        const int kvh = n >> 6, hd = n & 63;
        outb[((size_t)kvh * HEAD_DIM + hd) * (size_t)M + m] = f2bf(val);
      }
    }
  }
}

// --------------------------- flash attention -------------------------------
// Grid: (S/64, NH). Block: 128 threads = 4 waves; wave owns 16 queries.
// Qb: [S][HID] bf16 (post-RoPE), Kb: [S][KV_DIM] bf16 (post-RoPE),
// Vt: [NKV][HD][S] bf16, AO: [S][HID] bf16.

__global__ __launch_bounds__(128) void k_attn(const unsigned short* __restrict__ Qb,
                                              const unsigned short* __restrict__ Kb,
                                              const unsigned short* __restrict__ Vt,
                                              unsigned short* __restrict__ AO) {
  __shared__ __align__(16) unsigned short pl[4][16 * 32];

  const int head = blockIdx.y;
  const int kvh  = head >> 2;               // NH / NKV = 4
  const int wave = threadIdx.x >> 5;
  const int lane = threadIdx.x & 31;
  const int ln   = lane & 15;
  const int hh   = lane >> 4;
  const int q0   = blockIdx.x * 64 + wave * 16;

  // Q: two 16x32 A-fragments (head dim 64)
  const unsigned short* qrow = Qb + (size_t)(q0 + ln) * HID_DIM + head * HEAD_DIM;
  v16bf qa[2];
#pragma unroll
  for (int d = 0; d < 2; ++d) {
    const int base = d * 32 + (hh ? 16 : 0);
    qa[d] = ld16(qrow + base, qrow + base + 8);
  }

  const v8f z8 = {0.f, 0.f, 0.f, 0.f, 0.f, 0.f, 0.f, 0.f};
  v8f o[4] = {z8, z8, z8, z8};
  float mrow[8], lrow[8];
#pragma unroll
  for (int i = 0; i < 8; ++i) { mrow[i] = -3.0e38f; lrow[i] = 0.f; }

  int kstart = q0 - (WIN - 1);
  if (kstart < 0) kstart = 0;
  kstart &= ~31;
  const int kend = q0 + 15;
  unsigned short* pw = pl[wave];
  const float NEG_INF = -__builtin_huge_valf();

  for (int kb = kstart; kb <= kend; kb += 32) {
    // ---- scores S = Q @ K^T  (16 queries x 32 keys) ----
    v8f sc[2];
#pragma unroll
    for (int s2 = 0; s2 < 2; ++s2) {
      const unsigned short* krow =
          Kb + (size_t)(kb + s2 * 16 + ln) * KV_DIM + kvh * HEAD_DIM;
      v8f c = z8;
#pragma unroll
      for (int d = 0; d < 2; ++d) {
        const int base = d * 32 + hh * 16;
        v16bf kf = ld16(krow + base, krow + base + 8);
        c = wmma_bf16(qa[d], kf, c);
      }
      sc[s2] = c;
    }

    // ---- scale + sliding-window causal mask ----
#pragma unroll
    for (int s2 = 0; s2 < 2; ++s2)
#pragma unroll
      for (int i = 0; i < 8; ++i) {
        const int qi = q0 + i + 8 * hh;
        const int ki = kb + s2 * 16 + ln;
        float v = sc[s2][i] * 0.125f;          // 1/sqrt(64)
        if (ki > qi || ki < qi - (WIN - 1)) v = NEG_INF;
        sc[s2][i] = v;
      }

    // ---- online softmax ----
    float alpha[8];
#pragma unroll
    for (int i = 0; i < 8; ++i) {
      const float t = rmax16(fmaxf(sc[0][i], sc[1][i]));
      const float mn = fmaxf(mrow[i], t);
      alpha[i] = __expf(mrow[i] - mn);
      mrow[i] = mn;
    }
#pragma unroll
    for (int s2 = 0; s2 < 2; ++s2)
#pragma unroll
      for (int i = 0; i < 8; ++i)
        sc[s2][i] = __expf(sc[s2][i] - mrow[i]);   // exp(-inf)=0 for masked
#pragma unroll
    for (int i = 0; i < 8; ++i) {
      const float rs = rsum16(sc[0][i] + sc[1][i]);
      lrow[i] = lrow[i] * alpha[i] + rs;
    }
#pragma unroll
    for (int f = 0; f < 4; ++f)
#pragma unroll
      for (int i = 0; i < 8; ++i) o[f][i] *= alpha[i];

    // ---- re-stripe P: C-layout -> A-layout via per-wave LDS tile ----
#pragma unroll
    for (int s2 = 0; s2 < 2; ++s2)
#pragma unroll
      for (int i = 0; i < 8; ++i)
        pw[(i + 8 * hh) * 32 + s2 * 16 + ln] = f2bf(sc[s2][i]);
    asm volatile("s_wait_dscnt 0" ::: "memory");
    const int cb = hh ? 16 : 0;
    v16bf pa = ld16(pw + ln * 32 + cb, pw + ln * 32 + cb + 8);

    // ---- O += P @ V  (V pre-transposed: column of B is contiguous) ----
#pragma unroll
    for (int f = 0; f < 4; ++f) {
      const unsigned short* vcol =
          Vt + ((size_t)kvh * HEAD_DIM + f * 16 + ln) * S_LEN + kb + hh * 16;
      v16bf vf = ld16(vcol, vcol + 8);
      o[f] = wmma_bf16(pa, vf, o[f]);
    }
  }

  // ---- normalize + store bf16 for the Wo GEMM ----
  float inv[8];
#pragma unroll
  for (int i = 0; i < 8; ++i) inv[i] = 1.0f / lrow[i];
#pragma unroll
  for (int f = 0; f < 4; ++f)
#pragma unroll
    for (int i = 0; i < 8; ++i) {
      const int tok = q0 + i + 8 * hh;
      AO[(size_t)tok * HID_DIM + head * HEAD_DIM + f * 16 + ln] =
          f2bf(o[f][i] * inv[i]);
    }
}

// ------------------------------- launch ------------------------------------

extern "C" void kernel_launch(void* const* d_in, const int* in_sizes, int n_in,
                              void* d_out, int out_size, void* d_ws, size_t ws_size,
                              hipStream_t stream) {
  (void)in_sizes; (void)n_in; (void)out_size; (void)ws_size;
  const float* X  = (const float*)d_in[0];
  const float* Wq = (const float*)d_in[1];
  const float* Wk = (const float*)d_in[2];
  const float* Wv = (const float*)d_in[3];
  const float* Wo = (const float*)d_in[4];
  float* out = (float*)d_out;

  char* ws = (char*)d_ws;
  const size_t MB = 1024u * 1024u;
  unsigned short* Xb  = (unsigned short*)(ws + 0);        // 8 MB
  unsigned short* Wqt = (unsigned short*)(ws + 8  * MB);  // 8 MB
  unsigned short* Wkt = (unsigned short*)(ws + 16 * MB);  // 2 MB
  unsigned short* Wvt = (unsigned short*)(ws + 18 * MB);  // 2 MB
  unsigned short* Wot = (unsigned short*)(ws + 20 * MB);  // 8 MB
  unsigned short* Qb  = (unsigned short*)(ws + 28 * MB);  // 8 MB
  unsigned short* Kb  = (unsigned short*)(ws + 36 * MB);  // 2 MB
  unsigned short* Vt  = (unsigned short*)(ws + 38 * MB);  // 2 MB
  unsigned short* AO  = (unsigned short*)(ws + 40 * MB);  // 8 MB  (total 48 MB)

  // bf16 conversions / transposes
  k_f32_to_bf16<<<(S_LEN * HID_DIM) / 256, 256, 0, stream>>>(X, Xb, S_LEN * HID_DIM);
  k_transpose_bf16<<<(HID_DIM * HID_DIM) / 256, 256, 0, stream>>>(Wq, Wqt, HID_DIM, HID_DIM);
  k_transpose_bf16<<<(HID_DIM * KV_DIM)  / 256, 256, 0, stream>>>(Wk, Wkt, HID_DIM, KV_DIM);
  k_transpose_bf16<<<(HID_DIM * KV_DIM)  / 256, 256, 0, stream>>>(Wv, Wvt, HID_DIM, KV_DIM);
  k_transpose_bf16<<<(HID_DIM * HID_DIM) / 256, 256, 0, stream>>>(Wo, Wot, HID_DIM, HID_DIM);

  // projections (RoPE fused into Q/K epilogue, V written transposed)
  k_gemm<1><<<dim3(S_LEN / 128, HID_DIM / 64), 256, 0, stream>>>(
      Xb, Wqt, nullptr, Qb, S_LEN, HID_DIM, HID_DIM);
  k_gemm<1><<<dim3(S_LEN / 128, KV_DIM / 64), 256, 0, stream>>>(
      Xb, Wkt, nullptr, Kb, S_LEN, KV_DIM, HID_DIM);
  k_gemm<2><<<dim3(S_LEN / 128, KV_DIM / 64), 256, 0, stream>>>(
      Xb, Wvt, nullptr, Vt, S_LEN, KV_DIM, HID_DIM);

  // sliding-window flash attention
  k_attn<<<dim3(S_LEN / 64, N_HEADS), 128, 0, stream>>>(Qb, Kb, Vt, AO);

  // output projection -> fp32
  k_gemm<0><<<dim3(S_LEN / 128, HID_DIM / 64), 256, 0, stream>>>(
      AO, Wot, out, nullptr, S_LEN, HID_DIM, HID_DIM);
}
